// HeteroGNN_63977832841245
// MI455X (gfx1250) — matrix operand order repaired and checked by reading, not compiled, not run on hardware
//
#include <hip/hip_runtime.h>
#include <hip/hip_bf16.h>

typedef __attribute__((ext_vector_type(16))) _Float16 v16h;
typedef __attribute__((ext_vector_type(8)))  _Float16 v8h;
typedef __attribute__((ext_vector_type(8)))  float    v8f;

#define EPS_BN 1e-5f

// ---------------------------------------------------------------------------
// WMMA GEMM: C[M,N] = A[M,K] * B[K,N] (+ bias[N] optional), f32 in/out,
// f16 multiply via v_wmma_f32_16x16x32_f16.
// Block tile 64x64 (M%64==0, N%64==0 guaranteed by caller), 4 waves in a
// 2x2 grid, each wave computes a 32x32 tile = 2x2 WMMAs per K-step with
// fragment reuse. LDS tiles are double-buffered (one barrier per K-step);
// global loads for tile t+1 are issued into registers before the WMMAs of
// tile t. Bs is stored transposed [n][k] so every fragment is a pair of
// contiguous 16-byte LDS reads (ds_load_b128).
// ---------------------------------------------------------------------------
__global__ __launch_bounds__(128)
void wmma_gemm(const float* __restrict__ A, const float* __restrict__ B,
               const float* __restrict__ bias, float* __restrict__ C,
               int M, int N, int K)
{
    __shared__ _Float16 As[2][64 * 32];   // [buf][m][k]
    __shared__ _Float16 Bs[2][64 * 32];   // [buf] transposed: [n][k]

    const unsigned tid  = threadIdx.x;
    const unsigned wave = tid >> 5;
    const unsigned lane = tid & 31;
    const unsigned m0   = blockIdx.x * 64;
    const unsigned n0   = blockIdx.y * 64;

    const unsigned wm    = wave >> 1;           // wave row in 2x2 grid
    const unsigned wn    = wave & 1;            // wave col
    const unsigned mrow  = lane & 15;
    const unsigned khalf = lane >> 4;
    const unsigned nlane = lane & 15;
    const bool     kvec  = (K & 3) == 0;

    v8f acc00 = {}, acc01 = {}, acc10 = {}, acc11 = {};
    float4 va[4], vb[4];

    // load one 64x32 A tile + 32x64 B tile into registers (8x global_load_b128)
    auto gload = [&](int k0) {
        if (kvec && (k0 + 32 <= K)) {
            #pragma unroll
            for (int j = 0; j < 4; ++j) {                 // A: 64 rows x 32 k
                unsigned i = tid + j * 128, r = i >> 3, q = i & 7;
                va[j] = *(const float4*)(A + (m0 + r) * (unsigned)K + (unsigned)k0 + q * 4);
            }
            #pragma unroll
            for (int j = 0; j < 4; ++j) {                 // B: 32 k x 64 n
                unsigned i = tid + j * 128, r = i >> 4, q = i & 15;
                vb[j] = *(const float4*)(B + ((unsigned)k0 + r) * (unsigned)N + n0 + q * 4);
            }
        } else {
            #pragma unroll
            for (int j = 0; j < 4; ++j) {                 // A tail: per-element guard
                unsigned i = tid + j * 128, r = i >> 3, q = i & 7;
                const float* ap = A + (m0 + r) * (unsigned)K;
                int kb = k0 + (int)(q * 4);
                float4 t;
                t.x = (kb + 0 < K) ? ap[kb + 0] : 0.f;
                t.y = (kb + 1 < K) ? ap[kb + 1] : 0.f;
                t.z = (kb + 2 < K) ? ap[kb + 2] : 0.f;
                t.w = (kb + 3 < K) ? ap[kb + 3] : 0.f;
                va[j] = t;
            }
            #pragma unroll
            for (int j = 0; j < 4; ++j) {                 // B tail: row guard only
                unsigned i = tid + j * 128, r = i >> 4, q = i & 15;
                int k = k0 + (int)r;
                if (k < K) {
                    vb[j] = *(const float4*)(B + (unsigned)k * (unsigned)N + n0 + q * 4);
                } else {
                    float4 t; t.x = t.y = t.z = t.w = 0.f; vb[j] = t;
                }
            }
        }
    };
    // convert f32->f16 and store registers to LDS buffer b
    auto lstore = [&](int b) {
        #pragma unroll
        for (int j = 0; j < 4; ++j) {
            unsigned i = tid + j * 128, r = i >> 3, q = i & 7;
            _Float16* dp = As[b] + r * 32 + q * 4;
            dp[0] = (_Float16)va[j].x; dp[1] = (_Float16)va[j].y;
            dp[2] = (_Float16)va[j].z; dp[3] = (_Float16)va[j].w;
        }
        #pragma unroll
        for (int j = 0; j < 4; ++j) {
            unsigned i = tid + j * 128, r = i >> 4, q = i & 15;
            Bs[b][(q * 4 + 0) * 32 + r] = (_Float16)vb[j].x;
            Bs[b][(q * 4 + 1) * 32 + r] = (_Float16)vb[j].y;
            Bs[b][(q * 4 + 2) * 32 + r] = (_Float16)vb[j].z;
            Bs[b][(q * 4 + 3) * 32 + r] = (_Float16)vb[j].w;
        }
    };

    const int nk = (K + 31) >> 5;
    gload(0);
    lstore(0);
    __syncthreads();

    int buf = 0;
    for (int t = 0; t < nk; ++t) {
        const bool last = (t + 1 == nk);
        if (!last) gload((t + 1) * 32);      // prefetch next tile into registers

        // fragments per CDNA5 wave32 layouts (CDNA5 ISA §7.12.2)
        v16h af[2], bf[2];
        #pragma unroll
        for (int ms = 0; ms < 2; ++ms) {
            const _Float16* ap = As[buf] + (wm * 32 + ms * 16 + mrow) * 32;
            const v8h lo = *(const v8h*)(ap + 8 * khalf);        // K = 8*khalf+[0..7]
            const v8h hi = *(const v8h*)(ap + 16 + 8 * khalf);   // K = 16+8*khalf+[0..7]
            #pragma unroll
            for (int i = 0; i < 8; ++i) { af[ms][i] = lo[i]; af[ms][8 + i] = hi[i]; }
        }
        #pragma unroll
        for (int ns = 0; ns < 2; ++ns) {
            const _Float16* bp = Bs[buf] + (wn * 32 + ns * 16 + nlane) * 32;
            const v8h lo = *(const v8h*)(bp + 8 * khalf);        // K = 8*khalf+[0..7]
            const v8h hi = *(const v8h*)(bp + 16 + 8 * khalf);   // K = 16+8*khalf+[0..7]
            #pragma unroll
            for (int i = 0; i < 8; ++i) { bf[ns][i] = lo[i]; bf[ns][8 + i] = hi[i]; }
        }
        acc00 = __builtin_amdgcn_wmma_f32_16x16x32_f16(false, af[0], false, bf[0], (short)0, acc00, false, false);
        acc01 = __builtin_amdgcn_wmma_f32_16x16x32_f16(false, af[0], false, bf[1], (short)0, acc01, false, false);
        acc10 = __builtin_amdgcn_wmma_f32_16x16x32_f16(false, af[1], false, bf[0], (short)0, acc10, false, false);
        acc11 = __builtin_amdgcn_wmma_f32_16x16x32_f16(false, af[1], false, bf[1], (short)0, acc11, false, false);

        if (!last) {
            lstore(buf ^ 1);                 // fill other buffer
            __syncthreads();                 // one barrier per K-step
        }
        buf ^= 1;
    }

    // store 2x2 sub-tiles; C/D layout: lane holds (M = v + 8*khalf, N = lane&15)
    v8f* accs[4] = { &acc00, &acc01, &acc10, &acc11 };
    #pragma unroll
    for (int ms = 0; ms < 2; ++ms) {
        #pragma unroll
        for (int ns = 0; ns < 2; ++ns) {
            const unsigned ncol = n0 + wn * 32 + ns * 16 + nlane;
            const float bv = bias ? bias[ncol] : 0.f;
            const v8f& a = *accs[ms * 2 + ns];
            #pragma unroll
            for (int v = 0; v < 8; ++v) {
                unsigned m = m0 + wm * 32 + ms * 16 + v + 8 * khalf;
                C[m * (unsigned)N + ncol] = a[v] + bv;
            }
        }
    }
}

// ---------------------------------------------------------------------------
// edge features: f = leaky(si[src]+sj[dst]+fe+b), logits[h] = <f_h, attn_h>
// one block per edge, blockDim = H*De (64 or 256)
// ---------------------------------------------------------------------------
__global__ void egat_edge(const float* __restrict__ si, const float* __restrict__ sj,
                          const float* __restrict__ fe, const float* __restrict__ bias,
                          const float* __restrict__ attn,
                          const int* __restrict__ src, const int* __restrict__ dst,
                          float* __restrict__ f, float* __restrict__ logits,
                          int HD, int De)
{
    __shared__ float red[256];
    const int e = blockIdx.x;
    const int c = threadIdx.x;
    const int s = src[e], d = dst[e];
    float v = si[s * HD + c] + sj[d * HD + c] + fe[e * HD + c] + bias[c];
    v = (v > 0.f) ? v : 0.01f * v;            // leaky_relu(0.01)
    f[e * HD + c] = v;
    red[c] = v * attn[c];
    __syncthreads();
    if (c < 2) {                              // H == 2
        float sacc = 0.f;
        for (int i = 0; i < De; ++i) sacc += red[c * De + i];
        logits[e * 2 + c] = sacc;
    }
}

// ordered-uint encoding for float atomicMax
__device__ __forceinline__ unsigned fenc(float x) {
    unsigned u = __float_as_uint(x);
    return (u & 0x80000000u) ? ~u : (u | 0x80000000u);
}
__device__ __forceinline__ float fdec(unsigned u) {
    return (u & 0x80000000u) ? __uint_as_float(u & 0x7fffffffu)
                             : __uint_as_float(~u);
}

__global__ void seg_max(const float* __restrict__ logits, const int* __restrict__ dst,
                        unsigned* __restrict__ umax, int E)
{
    for (int i = blockIdx.x * blockDim.x + threadIdx.x; i < E * 2;
         i += gridDim.x * blockDim.x) {
        int e = i >> 1, h = i & 1;
        atomicMax(&umax[dst[e] * 2 + h], fenc(logits[i]));
    }
}

__global__ void seg_expsum(const float* __restrict__ logits, const int* __restrict__ dst,
                           const unsigned* __restrict__ umax, float* __restrict__ ex,
                           float* __restrict__ sum, int E)
{
    for (int i = blockIdx.x * blockDim.x + threadIdx.x; i < E * 2;
         i += gridDim.x * blockDim.x) {
        int e = i >> 1, h = i & 1;
        float m = fdec(umax[dst[e] * 2 + h]);
        float v = __expf(logits[i] - m);
        ex[i] = v;
        atomicAdd(&sum[dst[e] * 2 + h], v);
    }
}

__global__ void seg_norm(float* __restrict__ a, const int* __restrict__ dst,
                         const float* __restrict__ sum, int E)
{
    for (int i = blockIdx.x * blockDim.x + threadIdx.x; i < E * 2;
         i += gridDim.x * blockDim.x) {
        int e = i >> 1, h = i & 1;
        a[i] = a[i] / sum[dst[e] * 2 + h];
    }
}

// hout[dst] += a * hn[src]  (per head/channel); Dn = 1<<dn_sh
__global__ void egat_scatter(const float* __restrict__ a, const float* __restrict__ hn,
                             const int* __restrict__ src, const int* __restrict__ dst,
                             float* __restrict__ hout, unsigned total, int dn_sh)
{
    const unsigned dmask = (1u << dn_sh) - 1u;
    for (unsigned i = blockIdx.x * blockDim.x + threadIdx.x; i < total;
         i += gridDim.x * blockDim.x) {
        unsigned d = i & dmask;
        unsigned t = i >> dn_sh;
        unsigned h = t & 1u;
        unsigned e = t >> 1;
        float w = a[e * 2 + h];
        unsigned hd = (h << dn_sh) + d;
        atomicAdd(&hout[((unsigned)dst[e] << (dn_sh + 1)) + hd],
                  w * hn[((unsigned)src[e] << (dn_sh + 1)) + hd]);
    }
}

// column-wise mean / inv-std (biased variance, as torch BN train-mode)
__global__ void bn_stats(const float* __restrict__ x, int rows, int cols,
                         float* __restrict__ mean, float* __restrict__ istd)
{
    __shared__ float ssum[256], ssq[256];
    const int j = blockIdx.x;
    float s = 0.f, q = 0.f;
    for (int r = threadIdx.x; r < rows; r += blockDim.x) {
        float v = x[r * cols + j];
        s += v; q += v * v;
    }
    ssum[threadIdx.x] = s; ssq[threadIdx.x] = q;
    __syncthreads();
    for (int off = 128; off > 0; off >>= 1) {
        if (threadIdx.x < off) {
            ssum[threadIdx.x] += ssum[threadIdx.x + off];
            ssq[threadIdx.x]  += ssq[threadIdx.x + off];
        }
        __syncthreads();
    }
    if (threadIdx.x == 0) {
        float m = ssum[0] / rows;
        float v = ssq[0] / rows - m * m;
        mean[j] = m;
        istd[j] = rsqrtf(v + EPS_BN);
    }
}

// cols is a power of two: j = i & (cols-1)
__global__ void bn_apply(const float* __restrict__ x, const float* __restrict__ mean,
                         const float* __restrict__ istd, float* __restrict__ y,
                         unsigned n, unsigned cmask)
{
    for (unsigned i = blockIdx.x * blockDim.x + threadIdx.x; i < n;
         i += gridDim.x * blockDim.x) {
        unsigned j = i & cmask;
        y[i] = (x[i] - mean[j]) * istd[j];
    }
}

__global__ void vadd(const float* __restrict__ a, const float* __restrict__ b,
                     float* __restrict__ y, unsigned n)
{
    for (unsigned i = blockIdx.x * blockDim.x + threadIdx.x; i < n;
         i += gridDim.x * blockDim.x)
        y[i] = a[i] + b[i];
}

// (n, 2, D) -> (n, D) mean over heads, D = 1<<d_sh
__global__ void head_mean(const float* __restrict__ in, float* __restrict__ out,
                          unsigned total, int d_sh)
{
    const unsigned dmask = (1u << d_sh) - 1u;
    for (unsigned i = blockIdx.x * blockDim.x + threadIdx.x; i < total;
         i += gridDim.x * blockDim.x) {
        unsigned r = i >> d_sh, d = i & dmask;
        unsigned base = (r << (d_sh + 1)) + d;
        out[i] = 0.5f * (in[base] + in[base + (1u << d_sh)]);
    }
}

__global__ void fill_val(float* __restrict__ y, float v, unsigned n)
{
    for (unsigned i = blockIdx.x * blockDim.x + threadIdx.x; i < n;
         i += gridDim.x * blockDim.x)
        y[i] = v;
}

// ---------------------------------------------------------------------------
// host orchestration
// ---------------------------------------------------------------------------
namespace {
struct EgatP { const float *w_ni, *w_nj, *w_fij, *w_node, *attn, *b; };

constexpr int NREC  = 9600;
constexpr int NLIG  = 960;
constexpr int EREC  = 96000;
constexpr int ELIG  = 3840;
constexpr int ECPLX = 19200;
}

extern "C" void kernel_launch(void* const* d_in, const int* in_sizes, int n_in,
                              void* d_out, int out_size, void* d_ws, size_t ws_size,
                              hipStream_t stream)
{
    (void)in_sizes; (void)n_in; (void)out_size; (void)ws_size;

    // ---- inputs (setup_inputs order) ----
    const float* rec_n = (const float*)d_in[0];
    const float* rec_e = (const float*)d_in[1];
    const float* lig_n = (const float*)d_in[2];
    const float* lig_e = (const float*)d_in[3];
    const float* fw_e  = (const float*)d_in[4];
    const float* rv_e  = (const float*)d_in[5];
    const int* rec_src = (const int*)d_in[6];
    const int* rec_dst = (const int*)d_in[7];
    const int* lig_src = (const int*)d_in[8];
    const int* lig_dst = (const int*)d_in[9];
    const int* fw_src  = (const int*)d_in[10];
    const int* fw_dst  = (const int*)d_in[11];
    const int* rv_src  = (const int*)d_in[12];
    const int* rv_dst  = (const int*)d_in[13];

    auto ep = [&](int base) -> EgatP {
        return { (const float*)d_in[base + 0], (const float*)d_in[base + 1],
                 (const float*)d_in[base + 2], (const float*)d_in[base + 3],
                 (const float*)d_in[base + 4], (const float*)d_in[base + 5] };
    };
    EgatP Prec1 = ep(14), Plig1 = ep(20), Pfw1 = ep(26), Prv1 = ep(32);
    EgatP Prec2 = ep(38), Plig2 = ep(44), Pfw2 = ep(50), Prv2 = ep(56);
    const float* mlp1_rw = (const float*)d_in[62]; const float* mlp1_rb = (const float*)d_in[63];
    const float* mlp1_lw = (const float*)d_in[64]; const float* mlp1_lb = (const float*)d_in[65];
    const float* mlp2_rw = (const float*)d_in[66]; const float* mlp2_rb = (const float*)d_in[67];
    const float* mlp2_lw = (const float*)d_in[68]; const float* mlp2_lb = (const float*)d_in[69];

    // ---- workspace bump allocator (256B aligned chunks) ----
    float* wsf = (float*)d_ws;
    size_t ws_off = 0;
    auto allocf = [&](size_t n) -> float* {
        float* p = wsf + ws_off;
        ws_off += (n + 63) & ~(size_t)63;
        return p;
    };

    auto gemm = [&](const float* A, const float* Bm, const float* bias, float* C,
                    int M, int N, int K) {
        dim3 g(M / 64, N / 64);
        wmma_gemm<<<g, 128, 0, stream>>>(A, Bm, bias, C, M, N, K);
    };
    auto bn = [&](float* x, int rows, int cols) {
        float* mean = allocf(cols);
        float* istd = allocf(cols);
        bn_stats<<<cols, 256, 0, stream>>>(x, rows, cols, mean, istd);
        unsigned n = (unsigned)rows * cols;
        int blocks = (int)((n + 255) / 256); if (blocks > 4096) blocks = 4096;
        bn_apply<<<blocks, 256, 0, stream>>>(x, mean, istd, x, n, (unsigned)(cols - 1));
    };

    // full EGAT layer: writes f (E x 2De), a (E x 2), hout (Ndst x 2Dn)
    auto egat = [&](const float* nsrc, int Nsrc, int Ksrc,
                    const float* ndst, int Ndst, int Kdst,
                    const float* ef, int E, int Ke,
                    const int* src, const int* dst,
                    const EgatP& p, int De, int Dn, int dn_sh,
                    float* f, float* a, float* hout) {
        const int HDe = 2 * De, HDn = 2 * Dn;
        float* si = allocf((size_t)Nsrc * HDe);
        float* sj = allocf((size_t)Ndst * HDe);
        float* fe = allocf((size_t)E * HDe);
        float* hn = allocf((size_t)Nsrc * HDn);
        float* lg = allocf((size_t)E * 2);
        unsigned* umax = (unsigned*)allocf((size_t)Ndst * 2);
        float* sums = allocf((size_t)Ndst * 2);

        gemm(nsrc, p.w_ni,   nullptr, si, Nsrc, HDe, Ksrc);
        gemm(ndst, p.w_nj,   nullptr, sj, Ndst, HDe, Kdst);
        gemm(ef,   p.w_fij,  nullptr, fe, E,    HDe, Ke);
        gemm(nsrc, p.w_node, nullptr, hn, Nsrc, HDn, Ksrc);

        egat_edge<<<E, HDe, 0, stream>>>(si, sj, fe, p.b, p.attn, src, dst, f, lg, HDe, De);

        hipMemsetAsync(umax, 0, (size_t)Ndst * 2 * sizeof(unsigned), stream);
        hipMemsetAsync(sums, 0, (size_t)Ndst * 2 * sizeof(float), stream);
        int sb = (E * 2 + 255) / 256;
        seg_max<<<sb, 256, 0, stream>>>(lg, dst, umax, E);
        seg_expsum<<<sb, 256, 0, stream>>>(lg, dst, umax, a, sums, E);
        seg_norm<<<sb, 256, 0, stream>>>(a, dst, sums, E);

        hipMemsetAsync(hout, 0, (size_t)Ndst * HDn * sizeof(float), stream);
        unsigned tot = (unsigned)E * HDn;
        int scb = (int)((tot + 255) / 256); if (scb > 8192) scb = 8192;
        egat_scatter<<<scb, 256, 0, stream>>>(a, hn, src, dst, hout, tot, dn_sh);
    };

    // ---- d_out layout ----
    float* out      = (float*)d_out;
    float* out_rec  = out;                                   // 9600*512
    float* out_lig  = out_rec + (size_t)NREC * 512;          // 960*512
    float* out_rm   = out_lig + (size_t)NLIG * 512;          // 9600
    float* out_lm   = out_rm + NREC;                         // 960
    float* out_arec = out_lm + NLIG;                         // 96000*2
    float* out_alig = out_arec + (size_t)EREC * 2;           // 3840*2
    float* out_afw  = out_alig + (size_t)ELIG * 2;           // 19200*2
    float* out_arv  = out_afw + (size_t)ECPLX * 2;           // 19200*2

    // =============== layer 1 ===============
    float* rec_f1 = allocf((size_t)EREC * 64);
    float* lig_f1 = allocf((size_t)ELIG * 64);
    float* fw_f1  = allocf((size_t)ECPLX * 64);
    float* rv_f1  = allocf((size_t)ECPLX * 64);
    float* a1r = allocf((size_t)EREC * 2);
    float* a1l = allocf((size_t)ELIG * 2);
    float* a1f = allocf((size_t)ECPLX * 2);
    float* a1v = allocf((size_t)ECPLX * 2);
    float* ho_rec1 = allocf((size_t)NREC * 256);
    float* ho_lig1 = allocf((size_t)NLIG * 256);
    float* ho_fw1  = allocf((size_t)NREC * 256);
    float* ho_rv1  = allocf((size_t)NLIG * 256);

    egat(rec_n, NREC, 100, rec_n, NREC, 100, rec_e, EREC, 1,
         rec_src, rec_dst, Prec1, 32, 128, 7, rec_f1, a1r, ho_rec1);
    egat(lig_n, NLIG, 7,   lig_n, NLIG, 7,   lig_e, ELIG, 11,
         lig_src, lig_dst, Plig1, 32, 128, 7, lig_f1, a1l, ho_lig1);
    egat(lig_n, NLIG, 7,   rec_n, NREC, 100, fw_e,  ECPLX, 5,
         fw_src, fw_dst, Pfw1, 32, 128, 7, fw_f1, a1f, ho_fw1);
    egat(rec_n, NREC, 100, lig_n, NLIG, 7,   rv_e,  ECPLX, 5,
         rv_src, rv_dst, Prv1, 32, 128, 7, rv_f1, a1v, ho_rv1);

    bn(ho_rec1, NREC, 256);  bn(rec_f1, EREC, 64);     // rn1, re1
    bn(ho_lig1, NLIG, 256);  bn(lig_f1, ELIG, 64);     // ln1, le1
    bn(ho_fw1,  NREC, 256);  bn(fw_f1,  ECPLX, 64);    // fn1, fe1
    bn(ho_rv1,  NLIG, 256);  bn(rv_f1,  ECPLX, 64);    // vn1, ve1

    // residual MLP 1
    float* t_r = allocf((size_t)NREC * 256);
    float* t_l = allocf((size_t)NLIG * 256);
    gemm(ho_fw1, mlp1_rw, mlp1_rb, t_r, NREC, 256, 256); bn(t_r, NREC, 256);
    gemm(ho_rv1, mlp1_lw, mlp1_lb, t_l, NLIG, 256, 256); bn(t_l, NLIG, 256);
    float* rn2 = allocf((size_t)NREC * 256);
    float* ln2 = allocf((size_t)NLIG * 256);
    {
        unsigned nr = NREC * 256u, nl = NLIG * 256u;
        vadd<<<(int)((nr + 255) / 256), 256, 0, stream>>>(ho_rec1, t_r, rn2, nr);
        vadd<<<(int)((nl + 255) / 256), 256, 0, stream>>>(ho_lig1, t_l, ln2, nl);
    }

    // =============== layer 2 ===============
    float* rec_f2 = allocf((size_t)EREC * 256);
    float* lig_f2 = allocf((size_t)ELIG * 256);
    float* fw_f2  = allocf((size_t)ECPLX * 256);
    float* rv_f2  = allocf((size_t)ECPLX * 256);
    float* ho_rec2 = allocf((size_t)NREC * 1024);
    float* ho_lig2 = allocf((size_t)NLIG * 1024);
    float* ho_fw2  = allocf((size_t)NREC * 1024);
    float* ho_rv2  = allocf((size_t)NLIG * 1024);

    egat(rn2, NREC, 256, rn2, NREC, 256, rec_f1, EREC, 64,
         rec_src, rec_dst, Prec2, 128, 512, 9, rec_f2, out_arec, ho_rec2);
    egat(ln2, NLIG, 256, ln2, NLIG, 256, lig_f1, ELIG, 64,
         lig_src, lig_dst, Plig2, 128, 512, 9, lig_f2, out_alig, ho_lig2);
    egat(ln2, NLIG, 256, rn2, NREC, 256, fw_f1, ECPLX, 64,
         fw_src, fw_dst, Pfw2, 128, 512, 9, fw_f2, out_afw, ho_fw2);
    egat(rn2, NREC, 256, ln2, NLIG, 256, rv_f1, ECPLX, 64,
         rv_src, rv_dst, Prv2, 128, 512, 9, rv_f2, out_arv, ho_rv2);

    float* rn3 = allocf((size_t)NREC * 512);
    float* ln3 = allocf((size_t)NLIG * 512);
    float* fn3 = allocf((size_t)NREC * 512);
    float* vn3 = allocf((size_t)NLIG * 512);
    {
        unsigned nr = NREC * 512u, nl = NLIG * 512u;
        int br = (int)((nr + 255) / 256), bl = (int)((nl + 255) / 256);
        head_mean<<<br, 256, 0, stream>>>(ho_rec2, rn3, nr, 9);
        head_mean<<<bl, 256, 0, stream>>>(ho_lig2, ln3, nl, 9);
        head_mean<<<br, 256, 0, stream>>>(ho_fw2,  fn3, nr, 9);
        head_mean<<<bl, 256, 0, stream>>>(ho_rv2,  vn3, nl, 9);
    }
    bn(rn3, NREC, 512); bn(ln3, NLIG, 512);
    bn(fn3, NREC, 512); bn(vn3, NLIG, 512);

    // residual MLP 2 -> final node features directly into d_out
    float* t2_r = allocf((size_t)NREC * 512);
    float* t2_l = allocf((size_t)NLIG * 512);
    gemm(fn3, mlp2_rw, mlp2_rb, t2_r, NREC, 512, 512); bn(t2_r, NREC, 512);
    gemm(vn3, mlp2_lw, mlp2_lb, t2_l, NLIG, 512, 512); bn(t2_l, NLIG, 512);
    {
        unsigned nr = NREC * 512u, nl = NLIG * 512u;
        vadd<<<(int)((nr + 255) / 256), 256, 0, stream>>>(rn3, t2_r, out_rec, nr);
        vadd<<<(int)((nl + 255) / 256), 256, 0, stream>>>(ln3, t2_l, out_lig, nl);
    }

    // masks (all True -> 1.0f)
    fill_val<<<(NREC + 255) / 256, 256, 0, stream>>>(out_rm, 1.0f, NREC);
    fill_val<<<(NLIG + 255) / 256, 256, 0, stream>>>(out_lm, 1.0f, NLIG);
}
